// EdgeGraphLayer_420906795685
// MI455X (gfx1250) — matrix operand order repaired and checked by reading, not compiled, not run on hardware
//
#include <hip/hip_runtime.h>

// ---------------------------------------------------------------------------
// EdgeGraphLayer on MI455X (gfx1250): bf16 WMMA GEMMs, async global->LDS
// weight staging (double-buffered), fused int-bit atomic scatter-max.
// ---------------------------------------------------------------------------

typedef __attribute__((ext_vector_type(16))) __bf16 v16bf;
typedef __attribute__((ext_vector_type(8)))  float  v8f;

#define N_NODES 50000
#define N_EDGES 800000
#define FEAT    256
#define EDIM    64
#define OUT     256
#define KU      (OUT + EDIM)   // 320
#define LDSPITCH 40            // padded row pitch (elems) to spread LDS banks

// round-to-nearest-even f32 -> bf16
__device__ __forceinline__ unsigned short f2bf(float f) {
    unsigned u = __float_as_uint(f);
    u += 0x7FFFu + ((u >> 16) & 1u);
    return (unsigned short)(u >> 16);
}

union Frag {                    // one 16x32 bf16 operand fragment per lane
    v16bf v;
    unsigned short us[16];
    uint4 q[2];
};

// async 16B copy global -> LDS (ASYNCcnt-tracked, no VGPR round-trip).
// Taking ptrtoint of the LDS destination pointer (a) yields the 32-bit LDS
// byte offset (low 32 bits of the shared aperture address) and (b) escapes
// the __shared__ object so the "memory" clobber is known to alias it --
// without this the compiler deletes the ds_load readers as undef.
__device__ __forceinline__ void async_copy_b128(void* lds_dst, const void* g) {
    unsigned off32 = (unsigned)(unsigned long long)(size_t)lds_dst;
    asm volatile("global_load_async_to_lds_b128 %0, %1, off"
                 :: "v"(off32), "v"(g) : "memory");
}
__device__ __forceinline__ void wait_async0() {
    asm volatile("s_wait_asynccnt 0x0" ::: "memory");
}

// ---------------------------------------------------------------------------
// Prep kernels
// ---------------------------------------------------------------------------

// W[k][OUT] f32 -> Wt[n][K] bf16 (transposed so a B-fragment is two runs of 8 contiguous K)
__global__ void transpose_w_kernel(const float* __restrict__ W,
                                   unsigned short* __restrict__ Wt, int K) {
    int idx = blockIdx.x * blockDim.x + threadIdx.x;
    if (idx < OUT * K) {
        int n = idx / K, k = idx - n * K;
        Wt[idx] = f2bf(W[(size_t)k * OUT + n]);
    }
}

// f32 -> bf16, vectorized x4
__global__ void cvt_bf16_kernel(const float* __restrict__ in,
                                unsigned short* __restrict__ out, int n4) {
    int i = blockIdx.x * blockDim.x + threadIdx.x;
    if (i < n4) {
        float4 f = ((const float4*)in)[i];
        ushort4 o;
        o.x = f2bf(f.x); o.y = f2bf(f.y); o.z = f2bf(f.z); o.w = f2bf(f.w);
        ((ushort4*)out)[i] = o;
    }
}

__global__ void fill_zero_kernel(uint4* __restrict__ p, int n16) {
    int i = blockIdx.x * blockDim.x + threadIdx.x;
    if (i < n16) p[i] = make_uint4(0u, 0u, 0u, 0u);
}

// hv += msg (msg holds float bit-patterns written via int atomics); refresh bf16 copy
__global__ void update_hv_kernel(float* __restrict__ hv,
                                 const int* __restrict__ msgbits,
                                 unsigned short* __restrict__ hvb, int n4) {
    int i = blockIdx.x * blockDim.x + threadIdx.x;
    if (i < n4) {
        float4 h = ((const float4*)hv)[i];
        int4   m = ((const int4*)msgbits)[i];
        h.x += __int_as_float(m.x); h.y += __int_as_float(m.y);
        h.z += __int_as_float(m.z); h.w += __int_as_float(m.w);
        ((float4*)hv)[i] = h;
        ushort4 o;
        o.x = f2bf(h.x); o.y = f2bf(h.y); o.z = f2bf(h.z); o.w = f2bf(h.w);
        ((ushort4*)hvb)[i] = o;
    }
}

// ---------------------------------------------------------------------------
// Dense node GEMM:  C[M][OUT] = A[M][K](bf16) @ Wt[OUT][K]^T + bias, opt. ReLU
// Block = 256 threads = 8 waves as 4 M-tiles (64 rows) x 2 N-halves; weight
// K-slices (32x256 bf16) async-staged to LDS, double-buffered. Row-guarded.
// ---------------------------------------------------------------------------
__global__ __launch_bounds__(256)
void node_gemm_kernel(const unsigned short* __restrict__ A,
                      const unsigned short* __restrict__ Wt,
                      const float* __restrict__ bias,
                      float* __restrict__ Cf,
                      unsigned short* __restrict__ Cbf,
                      int M, int K, int do_relu) {
    __shared__ unsigned short ldsB[2][OUT * LDSPITCH];   // 2 x 20 KB

    const int tid  = threadIdx.x;
    const int wave = tid >> 5;
    const int lane = tid & 31;
    const int hi   = lane >> 4;
    const int l15  = lane & 15;
    const int mt   = wave >> 1;
    const int nh   = wave & 1;

    const int row  = blockIdx.x * 64 + mt * 16 + l15;
    const int arow_idx = row < M ? row : M - 1;          // clamp tail reads
    const unsigned short* arow = A + (size_t)arow_idx * K;

    v8f acc[8] = {};
    const int nkt = K >> 5;

    // stage slice 0: thread tid copies Wt row tid (64 B) into padded LDS row
    {
        const char* sp = (const char*)(Wt + (size_t)tid * K);
        char* dp = (char*)(&ldsB[0][tid * LDSPITCH]);
        async_copy_b128(dp +  0, sp +  0);
        async_copy_b128(dp + 16, sp + 16);
        async_copy_b128(dp + 32, sp + 32);
        async_copy_b128(dp + 48, sp + 48);
    }

    for (int kt = 0; kt < nkt; ++kt) {
        wait_async0();                   // own async writes to buf[kt&1] done
        __syncthreads();                 // everyone's writes visible; prev reads done
        if (kt + 1 < nkt) {
            const char* sp = (const char*)(Wt + (size_t)tid * K + (kt + 1) * 32);
            char* dp = (char*)(&ldsB[(kt + 1) & 1][tid * LDSPITCH]);
            async_copy_b128(dp +  0, sp +  0);
            async_copy_b128(dp + 16, sp + 16);
            async_copy_b128(dp + 32, sp + 32);
            async_copy_b128(dp + 48, sp + 48);
        }

        Frag a;
        const int ao = kt * 32 + hi * 8;
        a.q[0] = *(const uint4*)(arow + ao);
        a.q[1] = *(const uint4*)(arow + ao + 16);

        const unsigned short* lb = &ldsB[kt & 1][0];
#pragma unroll
        for (int t = 0; t < 8; ++t) {
            const int n = nh * 128 + t * 16 + l15;
            Frag b;
            const unsigned short* brow = lb + n * LDSPITCH + hi * 8;
            b.q[0] = *(const uint4*)(brow);
            b.q[1] = *(const uint4*)(brow + 16);
            acc[t] = __builtin_amdgcn_wmma_f32_16x16x32_bf16(
                false, a.v, false, b.v, (short)0, acc[t], false, false);
        }
    }

#pragma unroll
    for (int t = 0; t < 8; ++t) {
        const int n = nh * 128 + t * 16 + l15;
        const float bv = bias[n];
#pragma unroll
        for (int r = 0; r < 8; ++r) {
            const int rm = blockIdx.x * 64 + mt * 16 + hi * 8 + r;
            if (rm < M) {
                float v = acc[t][r] + bv;
                if (do_relu) v = fmaxf(v, 0.0f);
                const size_t o = (size_t)rm * OUT + n;
                if (Cf)  Cf[o]  = v;
                if (Cbf) Cbf[o] = f2bf(v);
            }
        }
    }
}

// ---------------------------------------------------------------------------
// Edge GEMM + fused scatter-max:
//   m[e] = relu(concat(hv_bf16[src[e]], he[e]) @ Wt^T + b); msg[dst[e]] max= m[e]
// Block = 256 threads = 8 waves as 4 M-tiles (64 edges) x 2 N-halves; weight
// K-slices async-staged to LDS, double-buffered, shared by all 8 waves.
// ---------------------------------------------------------------------------
__global__ __launch_bounds__(256)
void edge_gemm_scatter_kernel(const unsigned short* __restrict__ hvb,
                              const float* __restrict__ he,
                              const int* __restrict__ src,
                              const int* __restrict__ dst,
                              const unsigned short* __restrict__ Wt,   // [OUT][320]
                              const float* __restrict__ bias,
                              int* __restrict__ msg) {                 // f32 bits
    __shared__ unsigned short ldsB[2][OUT * LDSPITCH];   // 2 x 20 KB

    const int tid  = threadIdx.x;
    const int wave = tid >> 5;
    const int lane = tid & 31;
    const int hi   = lane >> 4;
    const int l15  = lane & 15;
    const int mt   = wave >> 1;      // M-tile 0..3
    const int nh   = wave & 1;       // N-half 0..1

    const int e0 = blockIdx.x * 64 + mt * 16;
    const int e  = e0 + l15;
    const int s  = src[e];
    const unsigned short* hrow  = hvb + (size_t)s * OUT;
    const float*          herow = he  + (size_t)e * EDIM;

    v8f acc[8] = {};

    {
        const char* sp = (const char*)(Wt + (size_t)tid * KU);
        char* dp = (char*)(&ldsB[0][tid * LDSPITCH]);
        async_copy_b128(dp +  0, sp +  0);
        async_copy_b128(dp + 16, sp + 16);
        async_copy_b128(dp + 32, sp + 32);
        async_copy_b128(dp + 48, sp + 48);
    }

#pragma unroll
    for (int kt = 0; kt < 10; ++kt) {
        wait_async0();
        __syncthreads();
        if (kt + 1 < 10) {
            const char* sp = (const char*)(Wt + (size_t)tid * KU + (kt + 1) * 32);
            char* dp = (char*)(&ldsB[(kt + 1) & 1][tid * LDSPITCH]);
            async_copy_b128(dp +  0, sp +  0);
            async_copy_b128(dp + 16, sp + 16);
            async_copy_b128(dp + 32, sp + 32);
            async_copy_b128(dp + 48, sp + 48);
        }

        // ---- A fragment for this wave's M-tile ----
        Frag a;
        if (kt < 8) {                            // gathered node state (L2 resident)
            const int ao = kt * 32 + hi * 8;
            a.q[0] = *(const uint4*)(hrow + ao);
            a.q[1] = *(const uint4*)(hrow + ao + 16);
        } else {                                 // streamed edge features (f32->bf16)
            const int ao = (kt - 8) * 32 + hi * 8;
            float4 f0 = *(const float4*)(herow + ao);
            float4 f1 = *(const float4*)(herow + ao + 4);
            float4 f2 = *(const float4*)(herow + ao + 16);
            float4 f3 = *(const float4*)(herow + ao + 20);
            a.us[0]  = f2bf(f0.x); a.us[1]  = f2bf(f0.y); a.us[2]  = f2bf(f0.z); a.us[3]  = f2bf(f0.w);
            a.us[4]  = f2bf(f1.x); a.us[5]  = f2bf(f1.y); a.us[6]  = f2bf(f1.z); a.us[7]  = f2bf(f1.w);
            a.us[8]  = f2bf(f2.x); a.us[9]  = f2bf(f2.y); a.us[10] = f2bf(f2.z); a.us[11] = f2bf(f2.w);
            a.us[12] = f2bf(f3.x); a.us[13] = f2bf(f3.y); a.us[14] = f2bf(f3.z); a.us[15] = f2bf(f3.w);
        }

        // ---- 8 N-tiles against LDS-resident B slice ----
        const unsigned short* lb = &ldsB[kt & 1][0];
#pragma unroll
        for (int t = 0; t < 8; ++t) {
            const int n = nh * 128 + t * 16 + l15;
            Frag b;
            const unsigned short* brow = lb + n * LDSPITCH + hi * 8;
            b.q[0] = *(const uint4*)(brow);
            b.q[1] = *(const uint4*)(brow + 16);
            acc[t] = __builtin_amdgcn_wmma_f32_16x16x32_bf16(
                false, a.v, false, b.v, (short)0, acc[t], false, false);
        }
    }

    // epilogue: bias + relu, then non-returning int-bit atomic max (values >= 0)
    int dvals[8];
#pragma unroll
    for (int r = 0; r < 8; ++r) dvals[r] = dst[e0 + hi * 8 + r];
#pragma unroll
    for (int t = 0; t < 8; ++t) {
        const int n = nh * 128 + t * 16 + l15;
        const float bv = bias[n];
#pragma unroll
        for (int r = 0; r < 8; ++r) {
            const float v = fmaxf(acc[t][r] + bv, 0.0f);
            atomicMax(msg + (size_t)dvals[r] * OUT + n, __float_as_int(v));
        }
    }
}

// ---------------------------------------------------------------------------
// Launcher
// ---------------------------------------------------------------------------
extern "C" void kernel_launch(void* const* d_in, const int* in_sizes, int n_in,
                              void* d_out, int out_size, void* d_ws, size_t ws_size,
                              hipStream_t stream) {
    const float* x    = (const float*)d_in[0];
    const float* he   = (const float*)d_in[1];
    const int*   src  = (const int*)d_in[2];
    const int*   dst  = (const int*)d_in[3];
    const float* W_in = (const float*)d_in[4];
    const float* b_in = (const float*)d_in[5];
    const float* W_u0 = (const float*)d_in[6];
    const float* b_u0 = (const float*)d_in[7];
    const float* W_u1 = (const float*)d_in[8];
    const float* b_u1 = (const float*)d_in[9];
    const float* W_ro = (const float*)d_in[10];
    const float* b_ro = (const float*)d_in[11];

    char* ws = (char*)d_ws;
    size_t off = 0;
    auto alloc = [&](size_t bytes) -> void* {
        void* p = ws + off;
        off += (bytes + 255) & ~(size_t)255;
        return p;
    };

    const size_t NE = (size_t)N_NODES * OUT;                 // 12.8M elems
    float*          hv   = (float*)alloc(NE * sizeof(float));          // 51.2 MB
    int*            msg  = (int*)alloc(NE * sizeof(int));              // 51.2 MB
    unsigned short* hvb  = (unsigned short*)alloc(NE * sizeof(short)); // 25.6 MB
    unsigned short* Wt_in = (unsigned short*)alloc((size_t)OUT * FEAT * 2);
    unsigned short* Wt_u0 = (unsigned short*)alloc((size_t)OUT * KU * 2);
    unsigned short* Wt_u1 = (unsigned short*)alloc((size_t)OUT * KU * 2);
    unsigned short* Wt_ro = (unsigned short*)alloc((size_t)OUT * OUT * 2);
    // x_bf16 aliases msg (msg is zeroed only after the input GEMM consumes it)
    unsigned short* xb = (unsigned short*)msg;

    // weight transposes (f32 -> bf16 [N][K])
    transpose_w_kernel<<<(OUT * FEAT + 255) / 256, 256, 0, stream>>>(W_in, Wt_in, FEAT);
    transpose_w_kernel<<<(OUT * KU + 255) / 256, 256, 0, stream>>>(W_u0, Wt_u0, KU);
    transpose_w_kernel<<<(OUT * KU + 255) / 256, 256, 0, stream>>>(W_u1, Wt_u1, KU);
    transpose_w_kernel<<<(OUT * OUT + 255) / 256, 256, 0, stream>>>(W_ro, Wt_ro, OUT);

    // x -> bf16
    {
        int n4 = (int)((size_t)N_NODES * FEAT / 4);
        cvt_bf16_kernel<<<(n4 + 255) / 256, 256, 0, stream>>>(x, xb, n4);
    }

    // hv = x @ W_in + b_in  (no relu); also emit bf16 copy
    const int node_blocks = (N_NODES + 63) / 64;
    node_gemm_kernel<<<node_blocks, 256, 0, stream>>>(xb, Wt_in, b_in, hv, hvb,
                                                      N_NODES, FEAT, 0);

    const int n16 = (int)(NE / 4);   // uint4 count for zero-fill
    const int n4  = (int)(NE / 4);   // float4 count for update

    // two message-passing rounds
    const unsigned short* Wts[2] = {Wt_u0, Wt_u1};
    const float* bs[2] = {b_u0, b_u1};
    for (int r = 0; r < 2; ++r) {
        fill_zero_kernel<<<(n16 + 255) / 256, 256, 0, stream>>>((uint4*)msg, n16);
        edge_gemm_scatter_kernel<<<N_EDGES / 64, 256, 0, stream>>>(
            hvb, he, src, dst, Wts[r], bs[r], msg);
        update_hv_kernel<<<(n4 + 255) / 256, 256, 0, stream>>>(hv, msg, hvb, n4);
    }

    // out = relu(hv @ W_ro + b_ro)
    node_gemm_kernel<<<node_blocks, 256, 0, stream>>>(hvb, Wt_ro, b_ro,
                                                      (float*)d_out, nullptr,
                                                      N_NODES, OUT, 1);
}
